// HeteroGNN_22548578304461
// MI455X (gfx1250) — matrix operand order repaired and checked by reading, not compiled, not run on hardware
//
#include <hip/hip_runtime.h>
#include <hip/hip_bf16.h>

// ---------------------------------------------------------------------------
// Problem constants (match reference)
// ---------------------------------------------------------------------------
#define N_P 100000
#define N_A 50000
#define E_C 1600000
#define E_W 800000
#define DIN 768
#define DH  128
#define WSZ (DH * DH)

typedef __attribute__((ext_vector_type(16))) __bf16 v16bf;
typedef __attribute__((ext_vector_type(8)))  float  v8f;

static __device__ __forceinline__ __bf16 f2bf(float f) { return (__bf16)f; }
static __device__ __forceinline__ float  bf2f(__bf16 h) { return (float)h; }

// Load a 16-element bf16 WMMA fragment from two contiguous 16-byte chunks.
static __device__ __forceinline__ v16bf load_frag(const __bf16* p0, const __bf16* p1) {
    union { v16bf v; uint4 u[2]; } r;
    r.u[0] = *(const uint4*)p0;
    r.u[1] = *(const uint4*)p1;
    return r.v;
}

// ---------------------------------------------------------------------------
// Edge-count kernel: cnt[dst] += 1.0  (computed once; reused by both layers)
// ---------------------------------------------------------------------------
__global__ void count_kernel(const int* __restrict__ dst, float* __restrict__ cnt, int nE) {
    int e = blockIdx.x * blockDim.x + threadIdx.x;
    if (e < nE) atomicAdd(&cnt[dst[e]], 1.0f);
}

// ---------------------------------------------------------------------------
// Scatter-add: one wave per edge; bf16 source rows (halved gather traffic),
// f32 atomic accumulation. 4 features per lane.
// ---------------------------------------------------------------------------
__global__ __launch_bounds__(256) void scatter_add_kernel(
    const __bf16* __restrict__ feat, const int* __restrict__ src,
    const int* __restrict__ dst, float* __restrict__ accum, int nE)
{
    int e = blockIdx.x * 8 + (threadIdx.x >> 5);
    if (e >= nE) return;
    int lane = threadIdx.x & 31;
    int s = src[e], d = dst[e];
    union { uint2 u; __bf16 h[4]; } v;
    v.u = ((const uint2*)(feat + (size_t)s * DH))[lane];
    float* o = accum + (size_t)d * DH + lane * 4;
    atomicAdd(o + 0, bf2f(v.h[0]));
    atomicAdd(o + 1, bf2f(v.h[1]));
    atomicAdd(o + 2, bf2f(v.h[2]));
    atomicAdd(o + 3, bf2f(v.h[3]));
}

// ---------------------------------------------------------------------------
// Mean-finalize: read f32 sums + counts, emit bf16 feature table for WMMA.
// ---------------------------------------------------------------------------
__global__ void finalize_mean_kernel(const float* __restrict__ agg, const float* __restrict__ cnt,
                                     __bf16* __restrict__ out, int n)
{
    int i = blockIdx.x * blockDim.x + threadIdx.x;
    if (i >= n * DH) return;
    float c = cnt[i >> 7];
    c = c < 1.0f ? 1.0f : c;
    out[i] = f2bf(agg[i] * (1.0f / c));
}

// ---------------------------------------------------------------------------
// Transpose + convert weights to bf16, N-major: Wt[n*K+k] = W0[k*N+n] (+W1[..])
// ---------------------------------------------------------------------------
__global__ void transpose_cvt_kernel(const float* __restrict__ W0, const float* __restrict__ W1,
                                     __bf16* __restrict__ Wt, int K, int N)
{
    int i = blockIdx.x * blockDim.x + threadIdx.x;
    if (i >= K * N) return;
    int n = i / K, k = i % K;
    float v = W0[(size_t)k * N + n];
    if (W1) v += W1[(size_t)k * N + n];
    Wt[i] = f2bf(v);
}

// bsum[l][c] = bl[l][0][c] + bl[l][1][c]
__global__ void bias_sum_kernel(const float* __restrict__ bl, float* __restrict__ bsum) {
    int i = blockIdx.x * blockDim.x + threadIdx.x;
    if (i >= 2 * DH) return;
    int l = i / DH, c = i % DH;
    bsum[i] = bl[(l * 3 + 0) * DH + c] + bl[(l * 3 + 1) * DH + c];
}

// ---------------------------------------------------------------------------
// Projection: out = relu(LayerNorm(x @ W + b) * g + bb)  -> bf16 feature table
// Block = 256 threads = 8 waves; tile = 16 rows x 128 cols; K = 768.
// A: f32 activations (wide float4 loads + v_cvt_pk), B: pre-transposed bf16.
// ---------------------------------------------------------------------------
__global__ __launch_bounds__(256) void proj_kernel(
    const float* __restrict__ x, const __bf16* __restrict__ Wt,   // Wt: [DH][DIN] bf16
    const float* __restrict__ b, const float* __restrict__ g,
    const float* __restrict__ bb, __bf16* __restrict__ out16)
{
    __shared__ float tile[16][DH + 4];
    __shared__ float s_mu[16], s_rs[16];

    const int wave   = threadIdx.x >> 5;
    const int lane   = threadIdx.x & 31;
    const int laneHi = lane >> 4;
    const int m      = lane & 15;
    const int rowBase = blockIdx.x * 16;
    const int colBase = wave * 16;
    const int row = rowBase + m;
    const int n   = colBase + m;

    const float*  arow = x  + (size_t)row * DIN;
    const __bf16* bcol = Wt + (size_t)n   * DIN;

    v8f acc = {};
    for (int kc = 0; kc < DIN; kc += 32) {
        // A fragment: lane holds contiguous K = laneHi*8..+7 and 16+laneHi*8..+7
        const float* pa = arow + kc + laneHi * 8;
        float4 a0 = *(const float4*)(pa);
        float4 a1 = *(const float4*)(pa + 4);
        float4 a2 = *(const float4*)(pa + 16);
        float4 a3 = *(const float4*)(pa + 20);
        v16bf af;
        af[0]  = f2bf(a0.x); af[1]  = f2bf(a0.y); af[2]  = f2bf(a0.z); af[3]  = f2bf(a0.w);
        af[4]  = f2bf(a1.x); af[5]  = f2bf(a1.y); af[6]  = f2bf(a1.z); af[7]  = f2bf(a1.w);
        af[8]  = f2bf(a2.x); af[9]  = f2bf(a2.y); af[10] = f2bf(a2.z); af[11] = f2bf(a2.w);
        af[12] = f2bf(a3.x); af[13] = f2bf(a3.y); af[14] = f2bf(a3.z); af[15] = f2bf(a3.w);
        // B fragment: lane holds K = laneHi*8..+7 and 16+laneHi*8..+7 of column n
        const __bf16* pb = bcol + kc + laneHi * 8;
        v16bf bfv = load_frag(pb, pb + 16);
        acc = __builtin_amdgcn_wmma_f32_16x16x32_bf16(false, af, false, bfv,
                                                      (short)0, acc, false, false);
    }

    // C/D layout: VGPR r -> (M = r + 8*laneHi, N = lane%16)
#pragma unroll
    for (int r = 0; r < 8; ++r) {
        int M = r + (laneHi << 3);
        int C = colBase + (lane & 15);
        tile[M][C] = acc[r] + b[C];
    }
    __syncthreads();

    if (threadIdx.x < 16) {
        int t = threadIdx.x;
        float mu = 0.0f;
        for (int c = 0; c < DH; ++c) mu += tile[t][c];
        mu *= (1.0f / DH);
        float var = 0.0f;
        for (int c = 0; c < DH; ++c) { float d = tile[t][c] - mu; var += d * d; }
        var *= (1.0f / DH);
        s_mu[t] = mu;
        s_rs[t] = rsqrtf(var + 1e-5f);
    }
    __syncthreads();

#pragma unroll
    for (int j = 0; j < 8; ++j) {
        int e = threadIdx.x * 8 + j;
        int r = e >> 7, c = e & (DH - 1);
        float v = (tile[r][c] - s_mu[r]) * s_rs[r] * g[c] + bb[c];
        v = v > 0.0f ? v : 0.0f;
        out16[(size_t)(rowBase + r) * DH + c] = f2bf(v);
    }
}

// ---------------------------------------------------------------------------
// Fused SAGE GEMM (all-bf16, contiguous b128 fragment loads):
//   acc = A0@W0 + A1@W1 + [A2@W2];  v = relu(scale*(acc + bias)) [+ resid]
// Block = 64 rows x 128 cols (4 row-tiles per block): each B fragment is
// loaded ONCE into 8 VGPRs and reused by 4 WMMAs (4x less weight traffic).
// Row-tile guards are block-uniform -> EXEC stays all-1s around WMMA.
// ---------------------------------------------------------------------------
__global__ __launch_bounds__(256) void sage_gemm_kernel(
    const __bf16* __restrict__ A0, const __bf16* __restrict__ W0,
    const __bf16* __restrict__ A1, const __bf16* __restrict__ W1,
    const __bf16* __restrict__ A2, const __bf16* __restrict__ W2,
    const float* __restrict__ bias, float scale,
    const __bf16* __restrict__ resid,
    float* __restrict__ out32, __bf16* __restrict__ out16, int nRows)
{
    const int wave   = threadIdx.x >> 5;
    const int lane   = threadIdx.x & 31;
    const int laneHi = lane >> 4;
    const int m      = lane & 15;
    const int blockRow = blockIdx.x * 64;
    const int colBase  = wave * 16;
    const int n = colBase + m;

    int nT = (nRows - blockRow) >> 4;     // valid 16-row tiles in this block
    if (nT > 4) nT = 4;

    const __bf16* As[3] = { A0, A1, A2 };
    const __bf16* Ws[3] = { W0, W1, W2 };

    v8f acc[4] = { {}, {}, {}, {} };
#pragma unroll
    for (int t = 0; t < 3; ++t) {
        const __bf16* A = As[t];
        const __bf16* W = Ws[t];
        if (A == nullptr) continue;                  // uniform branch
        const __bf16* bcol = W + (size_t)n * DH;     // W is N-major [DH][DH]
#pragma unroll
        for (int kc = 0; kc < DH; kc += 32) {
            const __bf16* pb = bcol + kc + laneHi * 8;
            v16bf bfv = load_frag(pb, pb + 16);      // one B frag, 4 WMMAs
#pragma unroll
            for (int rt = 0; rt < 4; ++rt) {
                if (rt < nT) {                       // block-uniform guard
                    const __bf16* pa = A + (size_t)(blockRow + rt * 16 + m) * DH
                                         + kc + laneHi * 8;
                    v16bf af = load_frag(pa, pa + 16);
                    acc[rt] = __builtin_amdgcn_wmma_f32_16x16x32_bf16(
                        false, af, false, bfv, (short)0, acc[rt], false, false);
                }
            }
        }
    }

    const float bv = bias[colBase + (lane & 15)];
#pragma unroll
    for (int rt = 0; rt < 4; ++rt) {
        if (rt < nT) {
#pragma unroll
            for (int r = 0; r < 8; ++r) {
                int M = r + (laneHi << 3);
                size_t oidx = (size_t)(blockRow + rt * 16 + M) * DH + colBase + (lane & 15);
                float v = (acc[rt][r] + bv) * scale;
                v = v > 0.0f ? v : 0.0f;             // relu before residual
                if (resid) v += bf2f(resid[oidx]);
                if (out32) out32[oidx] = v;
                if (out16) out16[oidx] = f2bf(v);
            }
        }
    }
}

// ---------------------------------------------------------------------------
// Host orchestration (graph-capture safe: only async memset + kernel launches)
// ---------------------------------------------------------------------------
extern "C" void kernel_launch(void* const* d_in, const int* in_sizes, int n_in,
                              void* d_out, int out_size, void* d_ws, size_t ws_size,
                              hipStream_t stream)
{
    (void)in_sizes; (void)n_in; (void)out_size; (void)ws_size;

    const float* x_p  = (const float*)d_in[0];
    const float* x_a  = (const float*)d_in[1];
    const float* pw_p = (const float*)d_in[2];
    const float* pb_p = (const float*)d_in[3];
    const float* g_p  = (const float*)d_in[4];
    const float* bb_p = (const float*)d_in[5];
    const float* pw_a = (const float*)d_in[6];
    const float* pb_a = (const float*)d_in[7];
    const float* g_a  = (const float*)d_in[8];
    const float* bb_a = (const float*)d_in[9];
    const float* Wl   = (const float*)d_in[10];  // [2,3,128,128]
    const float* bl   = (const float*)d_in[11];  // [2,3,128]
    const float* Wr   = (const float*)d_in[12];  // [2,3,128,128]
    const int* cites_src  = (const int*)d_in[13];
    const int* cites_dst  = (const int*)d_in[14];
    const int* writes_src = (const int*)d_in[15];
    const int* writes_dst = (const int*)d_in[16];
    const int* rev_src    = (const int*)d_in[17];
    const int* rev_dst    = (const int*)d_in[18];

    float* dout = (float*)d_out;   // [N_P*128 | N_A*128]

    // ---- workspace carve-up ----
    char* base = (char*)d_ws;
    size_t o = 0;
    auto alloc = [&](size_t bytes) { char* p = base + o; o = (o + bytes + 255) & ~(size_t)255; return p; };

    float* aggc = (float*)alloc((size_t)N_P * DH * 4);
    float* aggw = (float*)alloc((size_t)N_P * DH * 4);
    float* aggr = (float*)alloc((size_t)N_A * DH * 4);
    float* cntc = (float*)alloc((size_t)N_P * 4);
    float* cntw = (float*)alloc((size_t)N_P * 4);
    float* cntr = (float*)alloc((size_t)N_A * 4);
    float* bsum = (float*)alloc(2 * DH * 4);

    __bf16* hp_bf   = (__bf16*)alloc((size_t)N_P * DH * 2);
    __bf16* ha_bf   = (__bf16*)alloc((size_t)N_A * DH * 2);
    __bf16* zp1_bf  = (__bf16*)alloc((size_t)N_P * DH * 2);
    __bf16* za1_bf  = (__bf16*)alloc((size_t)N_A * DH * 2);
    __bf16* aggc_bf = (__bf16*)alloc((size_t)N_P * DH * 2);
    __bf16* aggw_bf = (__bf16*)alloc((size_t)N_P * DH * 2);
    __bf16* aggr_bf = (__bf16*)alloc((size_t)N_A * DH * 2);

    __bf16* wt_pp = (__bf16*)alloc((size_t)DIN * DH * 2);   // proj paper  [128][768]
    __bf16* wt_pa = (__bf16*)alloc((size_t)DIN * DH * 2);   // proj author [128][768]
    __bf16* wt_l0[2], *wt_l1[2], *wt_rs[2], *wt_l2[2], *wt_r2[2];
    for (int l = 0; l < 2; ++l) {
        wt_l0[l] = (__bf16*)alloc(WSZ * 2);
        wt_l1[l] = (__bf16*)alloc(WSZ * 2);
        wt_rs[l] = (__bf16*)alloc(WSZ * 2);
        wt_l2[l] = (__bf16*)alloc(WSZ * 2);
        wt_r2[l] = (__bf16*)alloc(WSZ * 2);
    }

    // ---- edge counts (same for both layers) ----
    hipMemsetAsync(cntc, 0, (size_t)N_P * 4, stream);
    hipMemsetAsync(cntw, 0, (size_t)N_P * 4, stream);
    hipMemsetAsync(cntr, 0, (size_t)N_A * 4, stream);
    count_kernel<<<(E_C + 255) / 256, 256, 0, stream>>>(cites_dst,  cntc, E_C);
    count_kernel<<<(E_W + 255) / 256, 256, 0, stream>>>(writes_dst, cntw, E_W);
    count_kernel<<<(E_W + 255) / 256, 256, 0, stream>>>(rev_dst,    cntr, E_W);

    // ---- weight prep: transpose+convert to bf16 N-major ----
    const int TPB = 256;
    transpose_cvt_kernel<<<(DIN * DH + TPB - 1) / TPB, TPB, 0, stream>>>(pw_p, nullptr, wt_pp, DIN, DH);
    transpose_cvt_kernel<<<(DIN * DH + TPB - 1) / TPB, TPB, 0, stream>>>(pw_a, nullptr, wt_pa, DIN, DH);
    for (int l = 0; l < 2; ++l) {
        const float* Wl_l = Wl + (size_t)l * 3 * WSZ;
        const float* Wr_l = Wr + (size_t)l * 3 * WSZ;
        transpose_cvt_kernel<<<(WSZ + TPB - 1) / TPB, TPB, 0, stream>>>(Wl_l + 0 * WSZ, nullptr,        wt_l0[l], DH, DH);
        transpose_cvt_kernel<<<(WSZ + TPB - 1) / TPB, TPB, 0, stream>>>(Wl_l + 1 * WSZ, nullptr,        wt_l1[l], DH, DH);
        transpose_cvt_kernel<<<(WSZ + TPB - 1) / TPB, TPB, 0, stream>>>(Wr_l + 0 * WSZ, Wr_l + 1 * WSZ, wt_rs[l], DH, DH);
        transpose_cvt_kernel<<<(WSZ + TPB - 1) / TPB, TPB, 0, stream>>>(Wl_l + 2 * WSZ, nullptr,        wt_l2[l], DH, DH);
        transpose_cvt_kernel<<<(WSZ + TPB - 1) / TPB, TPB, 0, stream>>>(Wr_l + 2 * WSZ, nullptr,        wt_r2[l], DH, DH);
    }
    bias_sum_kernel<<<1, 256, 0, stream>>>(bl, bsum);

    // ---- projections (WMMA GEMM + fused LayerNorm + ReLU -> bf16) ----
    proj_kernel<<<N_P / 16, 256, 0, stream>>>(x_p, wt_pp, pb_p, g_p, bb_p, hp_bf);
    proj_kernel<<<N_A / 16, 256, 0, stream>>>(x_a, wt_pa, pb_a, g_a, bb_a, ha_bf);

    const int GP = (N_P + 63) / 64;   // 64-row blocks, papers
    const int GA = (N_A + 63) / 64;   // 64-row blocks, authors

    // ---- two SAGE layers ----
    const __bf16* xp = hp_bf;
    const __bf16* xa = ha_bf;
    for (int l = 0; l < 2; ++l) {
        hipMemsetAsync(aggc, 0, (size_t)N_P * DH * 4, stream);
        hipMemsetAsync(aggw, 0, (size_t)N_P * DH * 4, stream);
        hipMemsetAsync(aggr, 0, (size_t)N_A * DH * 4, stream);

        scatter_add_kernel<<<(E_C + 7) / 8, 256, 0, stream>>>(xp, cites_src,  cites_dst,  aggc, E_C);
        scatter_add_kernel<<<(E_W + 7) / 8, 256, 0, stream>>>(xa, writes_src, writes_dst, aggw, E_W);
        scatter_add_kernel<<<(E_W + 7) / 8, 256, 0, stream>>>(xp, rev_src,    rev_dst,    aggr, E_W);

        finalize_mean_kernel<<<((size_t)N_P * DH + 255) / 256, 256, 0, stream>>>(aggc, cntc, aggc_bf, N_P);
        finalize_mean_kernel<<<((size_t)N_P * DH + 255) / 256, 256, 0, stream>>>(aggw, cntw, aggw_bf, N_P);
        finalize_mean_kernel<<<((size_t)N_A * DH + 255) / 256, 256, 0, stream>>>(aggr, cntr, aggr_bf, N_A);

        // paper: zp = relu(0.5*(aggC@Wl[l,0] + aggW@Wl[l,1] + xp@(Wr[l,0]+Wr[l,1]) + bl[l,0]+bl[l,1]))
        // author: za = relu(aggR@Wl[l,2] + xa@Wr[l,2] + bl[l,2])
        if (l == 0) {
            sage_gemm_kernel<<<GP, 256, 0, stream>>>(
                aggc_bf, wt_l0[l], aggw_bf, wt_l1[l], xp, wt_rs[l],
                bsum + l * DH, 0.5f, nullptr, nullptr, zp1_bf, N_P);
            sage_gemm_kernel<<<GA, 256, 0, stream>>>(
                aggr_bf, wt_l2[l], xa, wt_r2[l], nullptr, nullptr,
                bl + (l * 3 + 2) * DH, 1.0f, nullptr, nullptr, za1_bf, N_A);
            xp = zp1_bf;
            xa = za1_bf;
        } else {
            sage_gemm_kernel<<<GP, 256, 0, stream>>>(
                aggc_bf, wt_l0[l], aggw_bf, wt_l1[l], xp, wt_rs[l],
                bsum + l * DH, 0.5f, hp_bf, dout, nullptr, N_P);
            sage_gemm_kernel<<<GA, 256, 0, stream>>>(
                aggr_bf, wt_l2[l], xa, wt_r2[l], nullptr, nullptr,
                bl + (l * 3 + 2) * DH, 1.0f, ha_bf, dout + (size_t)N_P * DH, nullptr, N_A);
        }
    }
}
